// InterpretableAttentionLayer_52037823758872
// MI455X (gfx1250) — matrix operand order repaired and checked by reading, not compiled, not run on hardware
//
#include <hip/hip_runtime.h>
#include <hip/hip_bf16.h>
#include <math.h>

// ---------------------------------------------------------------------------
// CDNA5 WMMA types (wave32): v_wmma_f32_16x16x32_bf16
// ---------------------------------------------------------------------------
typedef __attribute__((ext_vector_type(16))) __bf16 v16bf;
typedef __attribute__((ext_vector_type(8)))  __bf16 v8bf;
typedef __attribute__((ext_vector_type(8)))  float  v8f;

__device__ __forceinline__ unsigned bf16u(float f) {
    unsigned u = __builtin_bit_cast(unsigned, f);
    return (u + 0x7FFFu + ((u >> 16) & 1u)) >> 16;   // round-to-nearest-even
}
__device__ __forceinline__ unsigned pack_bf2(float lo, float hi) {
    return (bf16u(hi) << 16) | (bf16u(lo) & 0xFFFFu);
}
__device__ __forceinline__ unsigned hashu(unsigned x) {
    x ^= x >> 16; x *= 0x7feb352dU;
    x ^= x >> 15; x *= 0x846ca68bU;
    x ^= x >> 16; return x;
}

#define MT 64
#define NT 64
#define KT 32
#define HR 40   // halves per LDS row: KT + 8 pad (keeps 16B alignment, no bank conflicts)

// fixed problem dims (compile-time for clean epilogue codegen)
#define CL 2048   // L = S
#define CE 64     // head dim
#define CH 16     // heads

__device__ __forceinline__ v16bf frag16(const unsigned short* row, int o0, int o1) {
    v8bf a = *(const v8bf*)(row + o0);
    v8bf b = *(const v8bf*)(row + o1);
    return __builtin_shufflevector(a, b, 0,1,2,3,4,5,6,7,8,9,10,11,12,13,14,15);
}

// ---------------------------------------------------------------------------
// Generic batched GEMM: C[M,N] = A[M,K] @ B + bias
//   BMODE 0: Bm stored N x K (weight, used as W^T)
//   BMODE 1: Bm stored K x N
//   OMODE 0: C row-major M x N
//   OMODE 1: (B,H,L,E) head-split: bb=m/CL, l=m%CL, h=n/CE, e=n%CE (shift/mask)
// Block: 256 threads = 8 waves; tile 64x64; wave (tm,tn) computes 16x32 via
// two v_wmma_f32_16x16x32_bf16 per 32-wide K chunk. bf16 conversion happens
// once at LDS staging time (packed 2xbf16 stores).
// Requires M%64==0, N%64==0, K%32==0 (true for all calls below).
// ---------------------------------------------------------------------------
template <int BMODE, int OMODE>
__global__ __launch_bounds__(256) void wmma_gemm_bf16(
    const float* __restrict__ A, long sAb,
    const float* __restrict__ Bm, long sBb,
    const float* __restrict__ bias,
    float* __restrict__ C, long sCb,
    int M, int N, int K)
{
    A  += (long)blockIdx.z * sAb;
    Bm += (long)blockIdx.z * sBb;
    C  += (long)blockIdx.z * sCb;

    const int m0   = blockIdx.y * MT;
    const int n0   = blockIdx.x * NT;
    const int tid  = threadIdx.x;
    const int lane = tid & 31;
    const int wave = tid >> 5;
    const int tm   = wave & 3;     // 0..3 -> M sub-tile
    const int tn   = wave >> 2;    // 0..1 -> 32-wide N strip
    const int lr   = lane & 15;
    const int hi   = lane >> 4;    // 0 or 1
    const int kb   = hi * 8;

    __shared__ unsigned short As[MT * HR];   // bf16, row stride HR halves
    __shared__ unsigned short Bs[NT * HR];   // bf16, [n][k]

    unsigned* Asw = (unsigned*)As;
    unsigned* Bsw = (unsigned*)Bs;

    v8f acc0 = {}, acc1 = {};

    for (int k0 = 0; k0 < K; k0 += KT) {
        // ---- stage A tile: convert to bf16 once, packed 2-per-uint stores
        for (int i = tid; i < MT * (KT / 2); i += 256) {
            int r = i >> 4, c2 = i & 15;
            const float2 v = *(const float2*)&A[(long)(m0 + r) * K + k0 + 2 * c2];
            Asw[r * (HR / 2) + c2] = pack_bf2(v.x, v.y);
        }
        // ---- stage B tile as Bs[n][k]
        if (BMODE == 0) {
            for (int i = tid; i < NT * (KT / 2); i += 256) {
                int n = i >> 4, c2 = i & 15;
                const float2 v = *(const float2*)&Bm[(long)(n0 + n) * K + k0 + 2 * c2];
                Bsw[n * (HR / 2) + c2] = pack_bf2(v.x, v.y);
            }
        } else {
            for (int i = tid; i < NT * KT; i += 256) {
                int kk = i >> 6, n = i & 63;   // consecutive n -> coalesced global
                Bs[n * HR + kk] = (unsigned short)bf16u(Bm[(long)(k0 + kk) * N + n0 + n]);
            }
        }
        __builtin_prefetch(&A[(long)(m0 + (tid & 63)) * K + k0 + KT], 0, 0);
        __syncthreads();

        // ---- fragments (ISA 7.12.2 layouts), vector LDS reads
        const unsigned short* arow = As + (tm * 16 + lr) * HR;
        v16bf af = frag16(arow, kb, 16 + kb);             // A 16x32

        const unsigned short* brow0 = Bs + (tn * 32 + lr) * HR;
        const unsigned short* brow1 = Bs + (tn * 32 + 16 + lr) * HR;
        v16bf bf0 = frag16(brow0, kb, 16 + kb);  // B 32x16, N-tile 0
        v16bf bf1 = frag16(brow1, kb, 16 + kb);  // B 32x16, N-tile 1

        acc0 = __builtin_amdgcn_wmma_f32_16x16x32_bf16(
                   false, af, false, bf0, (short)0, acc0, false, false);
        acc1 = __builtin_amdgcn_wmma_f32_16x16x32_bf16(
                   false, af, false, bf1, (short)0, acc1, false, false);
        __syncthreads();
    }

    // ---- epilogue. C/D layout: VGPR r -> M=r (lanes<16) / r+8 (lanes>=16)
#pragma unroll
    for (int r = 0; r < 8; ++r) {
        int m = m0 + tm * 16 + r + hi * 8;
#pragma unroll
        for (int nsub = 0; nsub < 2; ++nsub) {
            int n = n0 + tn * 32 + nsub * 16 + lr;
            float vout = (nsub ? acc1[r] : acc0[r]) + (bias ? bias[n] : 0.0f);
            if (OMODE == 0) {
                C[(long)m * N + n] = vout;
            } else {
                // (B,H,L,E): all compile-time shifts/masks
                unsigned bb = (unsigned)m / CL;
                unsigned l  = (unsigned)m % CL;
                unsigned h  = (unsigned)n / CE;
                unsigned e  = (unsigned)n % CE;
                unsigned idx = (((bb * CH + h) * CL) + l) * CE + e;
                C[idx] = vout;
            }
        }
    }
}

// ---------------------------------------------------------------------------
// ProbSparse measure: M[b,h,l] = max_j q.k_idx(l,j) - sum_j(q.k_idx(l,j))/S
// one wave32 per (b,h,l); lane j (<SK) owns one sampled key row.
// ---------------------------------------------------------------------------
__global__ __launch_bounds__(256) void prob_measure(
    const float* __restrict__ qh, const float* __restrict__ kh,
    float* __restrict__ Mmat, int L, int S, int E, int SK)
{
    int gw   = blockIdx.x * 8 + (threadIdx.x >> 5);   // bh*L + l
    int lane = threadIdx.x & 31;
    int wloc = threadIdx.x >> 5;
    int l  = gw & (L - 1);
    int bh = gw / L;

    __shared__ float qsh[8][64];
    const float* qrow = qh + ((long)bh * L + l) * E;
    qsh[wloc][lane]      = qrow[lane];
    qsh[wloc][lane + 32] = qrow[lane + 32];
    __syncthreads();

    float d = 0.f;
    if (lane < SK) {
        unsigned kidx = hashu((unsigned)(l * SK + lane)) & (unsigned)(S - 1);
        const float4* krow = (const float4*)(kh + ((long)bh * S + kidx) * E);
#pragma unroll
        for (int e4 = 0; e4 < 16; ++e4) {
            float4 kv = krow[e4];
            d += qsh[wloc][4*e4+0] * kv.x + qsh[wloc][4*e4+1] * kv.y
               + qsh[wloc][4*e4+2] * kv.z + qsh[wloc][4*e4+3] * kv.w;
        }
    }
    float mx = (lane < SK) ? d : -INFINITY;
    float sm = (lane < SK) ? d : 0.f;
#pragma unroll
    for (int off = 16; off; off >>= 1) {
        mx = fmaxf(mx, __shfl_xor(mx, off, 32));
        sm += __shfl_xor(sm, off, 32);
    }
    if (lane == 0) Mmat[gw] = mx - sm / (float)S;
}

// ---------------------------------------------------------------------------
// top-24 indices per (b,h); also count per (b,l) how many heads picked l
// ---------------------------------------------------------------------------
__global__ __launch_bounds__(256) void topk_u(
    const float* __restrict__ Mmat, int* __restrict__ Mtop,
    int* __restrict__ cnt, int H, int L, int U)
{
    int bh = blockIdx.x, tid = threadIdx.x;
    __shared__ float vals[2048];
    __shared__ float rv[256];
    __shared__ int   ri[256];
    for (int i = tid; i < L; i += 256) vals[i] = Mmat[(long)bh * L + i];
    __syncthreads();
    for (int t = 0; t < U; ++t) {
        float best = -INFINITY; int bi = 0;
        for (int i = tid; i < L; i += 256) {
            float v = vals[i];
            if (v > best) { best = v; bi = i; }
        }
        rv[tid] = best; ri[tid] = bi;
        __syncthreads();
        for (int s = 128; s; s >>= 1) {
            if (tid < s && rv[tid + s] > rv[tid]) { rv[tid] = rv[tid + s]; ri[tid] = ri[tid + s]; }
            __syncthreads();
        }
        if (tid == 0) {
            int w = ri[0];
            Mtop[bh * U + t] = w;
            vals[w] = -INFINITY;
            atomicAdd(&cnt[(bh / H) * L + w], 1);
        }
        __syncthreads();
    }
}

// attn_m base: rows not selected by a head contribute uniform 1/S
__global__ void attn_base(float* __restrict__ attn, const int* __restrict__ cnt,
                          int S, int H)
{
    int row = blockIdx.y;                                  // b*L + l
    int s   = blockIdx.x * blockDim.x + threadIdx.x;
    attn[(long)row * S + s] =
        (float)(H - cnt[row]) * (1.0f / ((float)H * (float)S));
}

// ---------------------------------------------------------------------------
// For each selected (b,h,u): full scores row, softmax, scatter-add /H into attn
// ---------------------------------------------------------------------------
__global__ __launch_bounds__(256) void topq_softmax_scatter(
    const float* __restrict__ qh, const float* __restrict__ kh,
    const int* __restrict__ Mtop, float* __restrict__ attn,
    int H, int L, int S, int E, int U, float scale)
{
    int blk = blockIdx.x;
    int u  = blk % U;
    int bh = blk / U;
    int b  = bh / H;
    int tid = threadIdx.x;
    int l = Mtop[bh * U + u];

    __shared__ float qrow[64];
    __shared__ float sc[2048];
    __shared__ float red[256];

    if (tid < 64) qrow[tid] = qh[((long)bh * L + l) * E + tid];
    __syncthreads();

    for (int s = tid; s < S; s += 256) {
        const float4* krow = (const float4*)(kh + ((long)bh * S + s) * E);
        float d = 0.f;
#pragma unroll
        for (int e4 = 0; e4 < 16; ++e4) {
            float4 kv = krow[e4];
            d += qrow[4*e4+0] * kv.x + qrow[4*e4+1] * kv.y
               + qrow[4*e4+2] * kv.z + qrow[4*e4+3] * kv.w;
        }
        sc[s] = d * scale;
    }
    __syncthreads();

    float m = -INFINITY;
    for (int s = tid; s < S; s += 256) m = fmaxf(m, sc[s]);
    red[tid] = m; __syncthreads();
    for (int st = 128; st; st >>= 1) {
        if (tid < st) red[tid] = fmaxf(red[tid], red[tid + st]);
        __syncthreads();
    }
    m = red[0]; __syncthreads();

    float z = 0.f;
    for (int s = tid; s < S; s += 256) { float e = __expf(sc[s] - m); sc[s] = e; z += e; }
    red[tid] = z; __syncthreads();
    for (int st = 128; st; st >>= 1) {
        if (tid < st) red[tid] += red[tid + st];
        __syncthreads();
    }
    float inv = 1.0f / (red[0] * (float)H);

    for (int s = tid; s < S; s += 256)
        atomicAdd(&attn[((long)b * L + l) * S + s], sc[s] * inv);
}

// ---------------------------------------------------------------------------
extern "C" void kernel_launch(void* const* d_in, const int* in_sizes, int n_in,
                              void* d_out, int out_size, void* d_ws, size_t ws_size,
                              hipStream_t stream)
{
    const float* q  = (const float*)d_in[0];
    const float* k  = (const float*)d_in[1];
    const float* v  = (const float*)d_in[2];
    /* d_in[3] = mask, unused by the reference computation */
    const float* Wq = (const float*)d_in[4];
    const float* bq = (const float*)d_in[5];
    const float* Wk = (const float*)d_in[6];
    const float* bk = (const float*)d_in[7];
    const float* Wv = (const float*)d_in[8];
    const float* bv = (const float*)d_in[9];
    const float* Wo = (const float*)d_in[10];
    const float* bo = (const float*)d_in[11];

    const int B = 2, L = 2048, S = 2048, D = 1024, H = 16, E = 64, U = 24, SK = 24;

    float* out_final = (float*)d_out;                 // (B,L,D)
    float* attn      = out_final + (long)B * L * D;   // (B,L,S)

    char* ws = (char*)d_ws;
    float* qh   = (float*)ws;  ws += (long)B * H * L * E * 4;  // (B,H,L,E)
    float* kh   = (float*)ws;  ws += (long)B * H * S * E * 4;  // (B,H,S,E)
    float* vp   = (float*)ws;  ws += (long)B * S * E * 4;      // (B,S,E)
    float* Mm   = (float*)ws;  ws += (long)B * H * L * 4;      // (B,H,L)
    float* outm = (float*)ws;  ws += (long)B * L * E * 4;      // (B,L,E)
    int*   Mtop = (int*)ws;    ws += (long)B * H * U * 4;
    int*   cnt  = (int*)ws;    ws += (long)B * L * 4;

    hipMemsetAsync(cnt, 0, (size_t)B * L * 4, stream);

    dim3 blk(256);

    // qh = (q @ Wq^T + bq) -> (B,H,L,E)
    wmma_gemm_bf16<0, 1><<<dim3(D / NT, (B * L) / MT, 1), blk, 0, stream>>>(
        q, 0, Wq, 0, bq, qh, 0, B * L, D, D);
    // kh = (k @ Wk^T + bk) -> (B,H,S,E)
    wmma_gemm_bf16<0, 1><<<dim3(D / NT, (B * S) / MT, 1), blk, 0, stream>>>(
        k, 0, Wk, 0, bk, kh, 0, B * S, D, D);
    // vp = v @ Wv^T + bv -> (B,S,E)
    wmma_gemm_bf16<0, 0><<<dim3(E / NT, (B * S) / MT, 1), blk, 0, stream>>>(
        v, 0, Wv, 0, bv, vp, 0, B * S, E, D);

    // sparsity measure + top-u selection
    prob_measure<<<dim3(B * H * L / 8), blk, 0, stream>>>(qh, kh, Mm, L, S, E, SK);
    topk_u<<<dim3(B * H), blk, 0, stream>>>(Mm, Mtop, cnt, H, L, U);

    // attn_m: uniform base + scattered softmax rows (never materialize B,H,L,S)
    attn_base<<<dim3(S / 256, B * L), blk, 0, stream>>>(attn, cnt, S, H);
    topq_softmax_scatter<<<dim3(B * H * U), blk, 0, stream>>>(
        qh, kh, Mtop, attn, H, L, S, E, U, 0.125f /* 1/sqrt(64) */);

    // out_mid = attn_m @ vp  (per batch; B-matrix stored K x N)
    wmma_gemm_bf16<1, 0><<<dim3(E / NT, L / MT, B), blk, 0, stream>>>(
        attn, (long)L * S, vp, (long)S * E, nullptr, outm, (long)L * E,
        L, E, S);

    // out = out_mid @ Wo^T + bo -> (B,L,D)
    wmma_gemm_bf16<0, 0><<<dim3(D / NT, (B * L) / MT, 1), blk, 0, stream>>>(
        outm, 0, Wo, 0, bo, out_final, 0, B * L, D, E);
}
